// BayesianAttentionModules_56023553409187
// MI455X (gfx1250) — matrix operand order repaired and checked by reading, not compiled, not run on hardware
//
#include <hip/hip_runtime.h>
#include <hip/hip_bf16.h>

typedef __attribute__((ext_vector_type(16))) _Float16 v16h;
typedef __attribute__((ext_vector_type(8)))  _Float16 v8h;
typedef __attribute__((ext_vector_type(8)))  float    v8f;
typedef __attribute__((ext_vector_type(4)))  float    v4f;

#define DIMV      64
#define LSEQ      200
#define LPAD      224      // 7 * 32 (context K-tiling)
#define NPSI_TILE 13       // 13*16 = 208 >= 200

#define INV_SQRT_D 0.125f               // 1/sqrt(64)
#define SIGMA_Q    0.1f
#define SOFT_SCL   0.0625f              // BETA / TAU = 0.25 / 4.0
#define KL_C0      1.8075850929940457f  // ln(10) + 0.01/2 - 0.5
#define KL_C1      0.5f                 // 1 / (2 * sigma_p^2)

__device__ __forceinline__ float wave_sum(float v) {
#pragma unroll
  for (int m = 16; m >= 1; m >>= 1) v += __shfl_xor(v, m, 32);
  return v;
}
__device__ __forceinline__ float wave_max(float v) {
#pragma unroll
  for (int m = 16; m >= 1; m >>= 1) v = fmaxf(v, __shfl_xor(v, m, 32));
  return v;
}

// 16 consecutive f32 -> v16h (keeps loads as 4x b128, cvt packs)
__device__ __forceinline__ v16h cvt16(v4f a, v4f b, v4f c, v4f d) {
  v8f lo = __builtin_shufflevector(a, b, 0, 1, 2, 3, 4, 5, 6, 7);
  v8f hi = __builtin_shufflevector(c, d, 0, 1, 2, 3, 4, 5, 6, 7);
  v8h l8 = __builtin_convertvector(lo, v8h);
  v8h h8 = __builtin_convertvector(hi, v8h);
  return __builtin_shufflevector(l8, h8, 0, 1, 2, 3, 4, 5, 6, 7,
                                 8, 9, 10, 11, 12, 13, 14, 15);
}

__global__ __launch_bounds__(256) void bayes_attn_main(
    const float* __restrict__ Q, const float* __restrict__ K,
    const float* __restrict__ V, const unsigned char* __restrict__ mask,
    const float* __restrict__ eps, float* __restrict__ out,
    float* __restrict__ wsSum, float* __restrict__ wsCnt, int Btot)
{
  __shared__ float sm[8][208];                       // psi per wave
  __shared__ __align__(16) _Float16 wh[8][LPAD];     // weights per wave (f16)

  const int wave = threadIdx.x >> 5;
  const int lane = threadIdx.x & 31;
  const int b    = blockIdx.x * 8 + wave;
  if (b >= Btot) return;                             // wave-uniform exit

  const size_t  kv = (size_t)b * (LSEQ * DIMV);
  const float*  Kb = K + kv;
  const float*  Vb = V + kv;
  const float*  Qb = Q + (size_t)b * DIMV;
  const unsigned char* mb = mask + (size_t)b * LSEQ;
  const float*  eb = eps + (size_t)b * LSEQ;

  const bool hi  = lane >= 16;
  const int  lh  = lane & 15;
  const int  kb0 = hi ? 16 : 0;    // A halves 0-7  -> K = kb0 + h
  const int  kb1 = hi ? 24 : 8;    // A halves 8-15 -> K = kb1 + (h-8)
  const int  kbB = hi ? 16 : 0;    // B halves 0-15 -> K = kbB + h

  // ---------------- phase 1: psi = (K Q) / sqrt(d) via WMMA -----------------
  // B operand: Q broadcast across all 16 N columns (16 contiguous f32).
  v16h bq0, bq1;
  {
    const float* qp = Qb + kbB;
    bq0 = cvt16(*(const v4f*)(qp +  0), *(const v4f*)(qp +  4),
                *(const v4f*)(qp +  8), *(const v4f*)(qp + 12));
    bq1 = cvt16(*(const v4f*)(qp + 32), *(const v4f*)(qp + 36),
                *(const v4f*)(qp + 40), *(const v4f*)(qp + 44));
  }

#pragma unroll 1
  for (int t = 0; t < NPSI_TILE; ++t) {
    const int l0   = t * 16;
    const int row  = l0 + lh;
    const int rowc = (row < LSEQ) ? row : (LSEQ - 1);   // clamp: no predication
    const float* rp = Kb + rowc * DIMV;
    __builtin_prefetch(Vb + rowc * DIMV + (hi ? 32 : 0), 0, 1);  // warm L2 for phase 3
    v8f acc = {};
#pragma unroll
    for (int cc = 0; cc < 2; ++cc) {
      const float* p = rp + cc * 32;
      v16h a = cvt16(*(const v4f*)(p + kb0), *(const v4f*)(p + kb0 + 4),
                     *(const v4f*)(p + kb1), *(const v4f*)(p + kb1 + 4));
      acc = __builtin_amdgcn_wmma_f32_16x16x32_f16(
          false, a, false, (cc ? bq1 : bq0), (short)0, acc, false, false);
    }
    // D layout: VGPR r, lanes 0-15 -> M=r; lanes 16-31 -> M=r+8.
    if (lh == 0) {
      const int base = l0 + (hi ? 8 : 0);
#pragma unroll
      for (int r = 0; r < 8; ++r) sm[wave][base + r] = acc[r] * INV_SQRT_D;
    }
  }

  // -------- phase 2: masked softmax((psi + sq*eps) * beta/tau), KL ----------
  // Fully branchless: unconditional clamped loads, mask folded in via selects.
  float sc[7], ps[7], okf[7];
  float lmax = -__builtin_inff();
#pragma unroll
  for (int i = 0; i < 7; ++i) {
    const int l  = lane + 32 * i;                    // 0..223
    const int lc = (l < LSEQ) ? l : (LSEQ - 1);      // clamp: loads unconditional
    const unsigned char mv = mb[lc];                 // unconditional u8 load
    const float ev = eb[lc];                         // unconditional b32 load
    const float p  = sm[wave][lc];                   // unconditional ds load
    const bool ok  = (l < LSEQ) && (mv != 0);
    float s = (p + SIGMA_Q * ev) * SOFT_SCL;         // unconditional math
    s = ok ? s : -__builtin_inff();                  // single v_cndmask
    sc[i] = s; ps[i] = p; okf[i] = ok ? 1.0f : 0.0f;
    lmax = fmaxf(lmax, s);
  }
  lmax = wave_max(lmax);
  // all-masked guard: keeps (s - lmax) = -inf -> exp = 0 instead of NaN
  lmax = (lmax == -__builtin_inff()) ? 0.0f : lmax;

  float e7[7];
  float lsum = 0.0f, klp = 0.0f, cnt = 0.0f;
#pragma unroll
  for (int i = 0; i < 7; ++i) {
    const float er = __expf(sc[i] - lmax);           // masked lanes: exp(-inf)=0
    e7[i] = er;
    lsum += er;
    klp  += okf[i] * (KL_C0 + KL_C1 * ps[i] * ps[i]);
    cnt  += okf[i];
  }
  lsum = wave_sum(lsum);
  klp  = wave_sum(klp);
  cnt  = wave_sum(cnt);
  const float inv = (lsum > 0.0f) ? (1.0f / lsum) : 0.0f;
#pragma unroll
  for (int i = 0; i < 7; ++i)
    wh[wave][lane + 32 * i] = (_Float16)(e7[i] * inv);   // all 224 written; >=200 are 0
  if (lane == 0) { wsSum[b] = klp; wsCnt[b] = cnt; }

  // ------------- phase 3: context = w^T V via WMMA (A row-bcast) ------------
  v8f accn[4];
#pragma unroll
  for (int n = 0; n < 4; ++n) accn[n] = (v8f){};

  const float* vcol = Vb + lh;     // this lane's V column (d = 16n + lh)
#pragma unroll 1
  for (int j = 0; j < 7; ++j) {
    const int l0 = 32 * j;
    // A[m,k] = w[l0+k] broadcast over rows: two aligned ds_load_b128
    v8h wlo = *(const v8h*)&wh[wave][l0 + kb0];
    v8h whi = *(const v8h*)&wh[wave][l0 + kb1];
    v16h aw = __builtin_shufflevector(wlo, whi, 0, 1, 2, 3, 4, 5, 6, 7,
                                      8, 9, 10, 11, 12, 13, 14, 15);
    const int lrow0 = l0 + kbB;
#pragma unroll
    for (int n = 0; n < 4; ++n) {
      v16h bv;                               // B[k,n'] = V[l0+k, 16n+n']
#pragma unroll
      for (int h = 0; h < 16; ++h) {
        const int l  = lrow0 + h;
        const int lc = (l < LSEQ) ? l : (LSEQ - 1);  // clamped; w==0 beyond L
        bv[h] = (_Float16)vcol[lc * DIMV + 16 * n];
      }
      accn[n] = __builtin_amdgcn_wmma_f32_16x16x32_f16(
          false, aw, false, bv, (short)0, accn[n], false, false);
    }
  }
  // all D rows identical; lanes 0-15, VGPR0 hold context[16n + lane]
  if (lane < 16) {
#pragma unroll
    for (int n = 0; n < 4; ++n)
      out[(size_t)b * DIMV + 16 * n + lane] = accn[n][0];
  }
}

__global__ __launch_bounds__(256) void bayes_attn_finalize(
    const float* __restrict__ wsSum, const float* __restrict__ wsCnt,
    float* __restrict__ out, int Btot)
{
  __shared__ float s1[256], s2[256];
  float a = 0.0f, c = 0.0f;
  for (int i = threadIdx.x; i < Btot; i += 256) { a += wsSum[i]; c += wsCnt[i]; }
  s1[threadIdx.x] = a; s2[threadIdx.x] = c;
  __syncthreads();
#pragma unroll
  for (int s = 128; s > 0; s >>= 1) {
    if ((int)threadIdx.x < s) {
      s1[threadIdx.x] += s1[threadIdx.x + s];
      s2[threadIdx.x] += s2[threadIdx.x + s];
    }
    __syncthreads();
  }
  if (threadIdx.x == 0) {
    const float n = fmaxf(s2[0], 1.0f);
    out[(size_t)Btot * DIMV] = s1[0] / n;
  }
}

extern "C" void kernel_launch(void* const* d_in, const int* in_sizes, int n_in,
                              void* d_out, int out_size, void* d_ws, size_t ws_size,
                              hipStream_t stream) {
  const float*         Q    = (const float*)d_in[0];
  const float*         K    = (const float*)d_in[1];
  const float*         V    = (const float*)d_in[2];
  const unsigned char* mask = (const unsigned char*)d_in[3];
  const float*         eps  = (const float*)d_in[4];
  float* out = (float*)d_out;

  const int Btot = in_sizes[0] / DIMV;     // 4096
  float* wsSum = (float*)d_ws;
  float* wsCnt = wsSum + Btot;

  const int blocks = (Btot + 7) / 8;       // 8 waves (8 batch rows) per block
  bayes_attn_main<<<blocks, 256, 0, stream>>>(Q, K, V, mask, eps, out,
                                              wsSum, wsCnt, Btot);
  bayes_attn_finalize<<<1, 256, 0, stream>>>(wsSum, wsCnt, out, Btot);
}
